// RecurrentProp_29085518528945
// MI455X (gfx1250) — matrix operand order repaired and checked by reading, not compiled, not run on hardware
//
#include <hip/hip_runtime.h>

typedef __attribute__((ext_vector_type(16))) _Float16 v16h;
typedef __attribute__((ext_vector_type(8)))  _Float16 v8h;
typedef __attribute__((ext_vector_type(8)))  float    v8f;
typedef __attribute__((ext_vector_type(4)))  float    v4f;

#define CC 256
#define HW 96          // H == W == 96, also the scan length
#define HWHW (HW * HW) // 9216

#if __has_builtin(__builtin_amdgcn_sched_group_barrier)
#define SGB(mask, n) __builtin_amdgcn_sched_group_barrier((mask), (n), 0)
#else
#define SGB(mask, n)
#endif
// sched_group_barrier masks: 0x008 = WMMA, 0x020 = VMEM read,
// 0x040 = VMEM write, 0x100 = DS read, 0x200 = DS write.

template <bool B> struct BoolC { static constexpr bool value = B; };

// Grid: 96 blocks = 4 directions x 24 row-blocks (24*16 = 384 = N*96 rows/dir).
// Block: 512 threads = 16 waves; wave wv owns output channels [16*wv, 16*wv+16).
// Per step: h_tile(16x256) = x_tile + h_prev(16x256) @ W(256x256) as
// 8 x v_wmma_f32_16x16x32_f16 per wave, x entering as the f32 C operand.
__launch_bounds__(512, 1)
__global__ void spatial_rnn4_kernel(const float* __restrict__ data,
                                    const float* __restrict__ weight,
                                    float* __restrict__ out)
{
    // double-buffered 16x256 f16 hidden state (16 KB total)
    __shared__ alignas(16) _Float16 hbuf[2][16 * CC];

    const int tid = threadIdx.x;
    const int wv  = tid >> 5;        // wave id 0..15 -> cout 16-block
    const int l   = tid & 31;        // lane
    const int lh  = l & 15;          // lane within half-wave
    const int hi  = l >> 4;          // 0 = lanes 0-15, 1 = lanes 16-31

    const int bid   = blockIdx.x;    // 0..95
    const int dir   = bid / 24;      // 0:W fwd, 1:W bwd, 2:H fwd, 3:H bwd
    const int rb    = bid % 24;      // 16-row block within this direction
    const int n     = rb / 6;        // batch (96 rows / 16 = 6 blocks per batch)
    const int pbase = (rb % 6) * 16; // base of non-scan spatial coordinate

    const bool scanW = (dir < 2);
    const bool fwd   = ((dir & 1) == 0);

    const int cout  = (wv << 4) + lh;       // this lane's channel column (0..255)
    const int moff  = hi << 3;              // C/D layout: row offset 0 or 8
    const int koffA = hi << 4;              // A layout: K base 0 or 16
    const int koffB = hi << 4;              // B layout: K base 0 or 16

    // ---- Preload B operand: this wave's 256x16 weight slab as 8 f16 32x16 tiles.
    // B layout (16-bit, 32x16): lane half holds K = koffB + j, N = lane&15.
    v16h Bt[8];
    #pragma unroll
    for (int k = 0; k < 8; ++k) {
        #pragma unroll
        for (int j = 0; j < 16; ++j) {
            const int cin = (k << 5) + koffB + j;
            Bt[k][j] = (_Float16)weight[cin * CC + cout];
        }
    }

    // ---- h[-1] = 0: zero the buffer read at step 0 (buffer 0).
    {
        const int base = tid * 8;                  // 512 threads x 8 = 4096 halves
        #pragma unroll
        for (int j = 0; j < 8; ++j) hbuf[0][base + j] = (_Float16)0.0f;
    }
    __syncthreads();

    // ---- Scan loop, specialized per direction class so each block runs a
    // straight-line body (dir is uniform per block; barriers stay legal).
    auto scan_body = [&](auto SC) {
        constexpr bool SW = decltype(SC)::value;
        const int qstride = SW ? 1 : HW;           // stride of the scan axis
        const int p0 = pbase + moff;
        const int sp = SW ? p0 * HW : p0;          // row stride: HW (W-scan) or 1
        const int inb  = (n * CC + cout) * HWHW + sp;
        const int outb = (n * 4 * CC + dir * CC + cout) * HWHW + sp;

        // x tile: 8 strided scalars (W-scan) or two b128 (H-scan, contiguous).
        auto load_x = [&](int qq) -> v8f {
            v8f r;
            if constexpr (SW) {
                #pragma unroll
                for (int i = 0; i < 8; ++i) r[i] = data[inb + i * HW + qq];
            } else {
                const v4f x0 = *(const v4f*)(data + inb + qq * HW);
                const v4f x1 = *(const v4f*)(data + inb + qq * HW + 4);
                r = __builtin_shufflevector(x0, x1, 0, 1, 2, 3, 4, 5, 6, 7);
            }
            return r;
        };

        const int dq = fwd ? 1 : -1;
        int q = fwd ? 0 : (HW - 1);
        v8f xcur = load_x(q);                      // prime the x pipeline

        #pragma unroll 1
        for (int s = 0; s < HW; ++s) {
            const int qoff = q * qstride;

            v8f acc0 = xcur;       // x enters as the WMMA C operand
            v8f acc1;
            #pragma unroll
            for (int i = 0; i < 8; ++i) acc1[i] = 0.0f;

            // A operand from LDS: row = lane&15; per chunk two contiguous
            // 8-half runs at K = 32k+koffA and +8 (two ds_load_b128 each).
            const _Float16* rbuf = &hbuf[s & 1][lh * CC];
            v16h a[8];
            #pragma unroll
            for (int k = 0; k < 8; ++k) {
                const v8h lo = *(const v8h*)(rbuf + (k << 5) + koffA);
                const v8h hh = *(const v8h*)(rbuf + (k << 5) + koffA + 8);
                a[k] = __builtin_shufflevector(lo, hh,
                    0, 1, 2, 3, 4, 5, 6, 7, 8, 9, 10, 11, 12, 13, 14, 15);
            }
            #pragma unroll
            for (int k = 0; k < 8; ++k) {
                if (k & 1)
                    acc1 = __builtin_amdgcn_wmma_f32_16x16x32_f16(
                        false, a[k], false, Bt[k], (short)0, acc1, false, false);
                else
                    acc0 = __builtin_amdgcn_wmma_f32_16x16x32_f16(
                        false, a[k], false, Bt[k], (short)0, acc0, false, false);
            }

            // Next scan position (uniform; keep it in SGPRs).
            int qn = q + dq;
            qn = (qn < 0) ? 0 : ((qn > HW - 1) ? (HW - 1) : qn);
            qn = __builtin_amdgcn_readfirstlane(qn);
            const v8f xnext = load_x(qn);          // prefetch over the WMMA tail

            // Combine accumulators; f32 output is a write-only stream -> NT.
            float hv[8];
            #pragma unroll
            for (int i = 0; i < 8; ++i) hv[i] = acc0[i] + acc1[i];

            _Float16* wbuf = &hbuf[(s + 1) & 1][0];
            #pragma unroll
            for (int i = 0; i < 8; ++i)
                wbuf[(moff + i) * CC + cout] = (_Float16)hv[i];

            if constexpr (SW) {
                #pragma unroll
                for (int i = 0; i < 8; ++i)
                    __builtin_nontemporal_store(hv[i], out + outb + i * HW + qoff);
            } else {
                v4f o0, o1;
                #pragma unroll
                for (int i = 0; i < 4; ++i) { o0[i] = hv[i]; o1[i] = hv[i + 4]; }
                __builtin_nontemporal_store(o0, (v4f*)(out + outb + qoff));
                __builtin_nontemporal_store(o1, (v4f*)(out + outb + qoff + 4));
            }

            // Schedule shape: issue ALL 16 ds_load_b128 first (so the waits
            // become a decrementing s_wait_dscnt pipeline instead of 0x0
            // stalls), then WMMAs interleaved with the x-prefetch VMEM reads,
            // then the h writeback and the streaming output stores.
            SGB(0x100, 16);                        // 16 DS reads
            if constexpr (SW) {
                #pragma unroll
                for (int k = 0; k < 8; ++k) { SGB(0x008, 1); SGB(0x020, 1); }
            } else {
                SGB(0x008, 3); SGB(0x020, 2); SGB(0x008, 5);
            }
            SGB(0x200, 8);                         // 8 DS writes (h for s+1)
            SGB(0x040, SW ? 8 : 2);                // streaming output stores

            // Publish h[s]: wait only on LDS stores (DScnt); global output
            // stores keep draining in the background.
            __asm__ volatile(
                "s_wait_dscnt 0\n\t"
                "s_barrier_signal -1\n\t"
                "s_barrier_wait -1" ::: "memory");

            xcur = xnext;
            q = qn;
        }
    };

    if (scanW) scan_body(BoolC<true>{});
    else       scan_body(BoolC<false>{});
}

extern "C" void kernel_launch(void* const* d_in, const int* in_sizes, int n_in,
                              void* d_out, int out_size, void* d_ws, size_t ws_size,
                              hipStream_t stream) {
    (void)in_sizes; (void)n_in; (void)out_size; (void)d_ws; (void)ws_size;
    const float* data   = (const float*)d_in[0];   // (4, 256, 96, 96) f32
    const float* weight = (const float*)d_in[1];   // (256, 256) f32
    float*       outp   = (float*)d_out;           // (4, 1024, 96, 96) f32
    spatial_rnn4_kernel<<<dim3(96), dim3(512), 0, stream>>>(data, weight, outp);
}
